// Head_78091095376015
// MI455X (gfx1250) — matrix operand order repaired and checked by reading, not compiled, not run on hardware
//
#include <hip/hip_runtime.h>
#include <hip/hip_bf16.h>

typedef __attribute__((ext_vector_type(16))) _Float16 v16h;
typedef __attribute__((ext_vector_type(8)))  float    v8f;
typedef __attribute__((ext_vector_type(4)))  int      v4i;

#define TSEQ   4096
#define BATCH  4
#define CEMB   512
#define HEAD   64

// LDS row pitches (halves): odd multiples of 8 halves (16B) -> bank-conflict-free
// column reads while keeping b128 alignment.
#define KPITCH 72   // K / V^T tile row pitch (64 data + 8 pad)
#define PPITCH 40   // P slab row pitch (32 data + 8 pad)

#ifndef __has_builtin
#define __has_builtin(x) 0
#endif
#if __has_builtin(__builtin_amdgcn_global_load_async_to_lds_b128)
#define HAVE_ASYNC_LDS 1
#else
#define HAVE_ASYNC_LDS 0
#endif

union V16U { v16h v; uint4 u[2]; _Float16 h[16]; };

// B-operand (32x16, 16-bit): lane holds column N=lane&15, 16 contiguous K values
// starting at kstart=(lane>=16)*16.
__device__ __forceinline__ v16h ldB16(const _Float16* p) {
    V16U r;
    const uint4* q = (const uint4*)p;
    r.u[0] = q[0];
    r.u[1] = q[1];
    return r.v;
}

// A-operand (16x32, 16-bit): lane holds row M=lane&15; elements 0-7 = K
// [kstart, kstart+8), elements 8-15 = K [kstart+8, +8), kstart=(lane>=16)*16.
__device__ __forceinline__ v16h ldA16(const _Float16* p0, const _Float16* p1) {
    V16U r;
    r.u[0] = *(const uint4*)p0;
    r.u[1] = *(const uint4*)p1;
    return r.v;
}

// 16-byte global -> LDS copy: async on gfx1250 (ASYNCcnt), else through VGPRs.
__device__ __forceinline__ void cp16_g2l(_Float16* dst, const _Float16* src) {
#if HAVE_ASYNC_LDS
    __builtin_amdgcn_global_load_async_to_lds_b128(
        (__attribute__((address_space(1))) v4i*)src,
        (__attribute__((address_space(3))) v4i*)dst, 0, 0);
#else
    *(uint4*)dst = *(const uint4*)src;
#endif
}

__device__ __forceinline__ void wait_async_le8() {
#if HAVE_ASYNC_LDS
    asm volatile("s_wait_asynccnt 0x8" ::: "memory");
#endif
}
__device__ __forceinline__ void wait_async_le0() {
#if HAVE_ASYNC_LDS
    asm volatile("s_wait_asynccnt 0x0" ::: "memory");
#endif
}

// ---------------------------------------------------------------------------
// Kernel 0: transpose + f16-convert weights: Wt[m][h][c] = W_m[c][h]
// ---------------------------------------------------------------------------
__global__ __launch_bounds__(256) void wt_kernel(
    const float* __restrict__ Wq, const float* __restrict__ Wk,
    const float* __restrict__ Wv, _Float16* __restrict__ Wt) {
    int idx = blockIdx.x * 256 + threadIdx.x;          // 0 .. 3*64*512-1
    int m = idx / (HEAD * CEMB);
    int r = idx - m * (HEAD * CEMB);
    int h = r / CEMB;
    int c = r - h * CEMB;
    const float* W = (m == 0) ? Wq : (m == 1) ? Wk : Wv;
    Wt[idx] = (_Float16)W[c * HEAD + h];
}

// ---------------------------------------------------------------------------
// Kernel 1: projections, single pass over x. Block = 128 thr = 4 waves; wave
// owns a 32-row x tile; each A tile feeds q,k,v accumulators (24 WMMA chains).
// ---------------------------------------------------------------------------
__global__ __launch_bounds__(128) void proj_kernel(
    const float* __restrict__ x, const _Float16* __restrict__ Wt,
    _Float16* __restrict__ qh, _Float16* __restrict__ kh,
    _Float16* __restrict__ vT) {
    const int lane  = threadIdx.x & 31;
    const int w     = threadIdx.x >> 5;
    const long long gq = (long long)blockIdx.x * 128 + (long long)w * 32;
    const int b     = (int)(gq / TSEQ);
    const int trow  = (int)(gq % TSEQ);

    const int arow   = lane & 15;
    const int khalf  = (lane >> 4) << 3;
    const int kstart = (lane >> 4) << 4;
    const int hi     = lane >> 4;
    const int n15    = lane & 15;

    const float* xrow = x + gq * CEMB;

    v8f acc[3][2][4];
    #pragma unroll
    for (int m = 0; m < 3; ++m)
        #pragma unroll
        for (int t = 0; t < 2; ++t)
            #pragma unroll
            for (int n = 0; n < 4; ++n)
                acc[m][t][n] = (v8f){0, 0, 0, 0, 0, 0, 0, 0};

    #pragma unroll 1   // keep VGPRs < 256: 192 accs + A + B + addresses
    for (int k0 = 0; k0 < CEMB; k0 += 32) {
        V16U A[2];
        #pragma unroll
        for (int t = 0; t < 2; ++t) {
            const float* ap = xrow + (long long)(t * 16 + arow) * CEMB + k0;
            union { float f[16]; float4 q4[4]; } xa;
            xa.q4[0] = *(const float4*)(ap + kstart);
            xa.q4[1] = *(const float4*)(ap + kstart + 4);
            xa.q4[2] = *(const float4*)(ap + kstart + 8);
            xa.q4[3] = *(const float4*)(ap + kstart + 12);
            #pragma unroll
            for (int e = 0; e < 16; ++e) A[t].h[e] = (_Float16)xa.f[e];
        }
        #pragma unroll
        for (int m = 0; m < 3; ++m) {
            const _Float16* Wm = Wt + (size_t)m * (HEAD * CEMB) + k0 + kstart;
            #pragma unroll
            for (int n = 0; n < 4; ++n) {
                v16h Bv = ldB16(Wm + (size_t)(n * 16 + n15) * CEMB);
                acc[m][0][n] = __builtin_amdgcn_wmma_f32_16x16x32_f16(
                    false, A[0].v, false, Bv, (short)0, acc[m][0][n], false, false);
                acc[m][1][n] = __builtin_amdgcn_wmma_f32_16x16x32_f16(
                    false, A[1].v, false, Bv, (short)0, acc[m][1][n], false, false);
            }
        }
    }

    _Float16* vb = vT + (size_t)b * HEAD * TSEQ;
    #pragma unroll
    for (int t = 0; t < 2; ++t)
        #pragma unroll
        for (int n = 0; n < 4; ++n)
            #pragma unroll
            for (int r = 0; r < 8; ++r) {
                const long long row = gq + t * 16 + r + hi * 8;
                qh[row * HEAD + n * 16 + n15] = (_Float16)acc[0][t][n][r];
                kh[row * HEAD + n * 16 + n15] = (_Float16)acc[1][t][n][r];
                vb[(size_t)(n * 16 + n15) * TSEQ + (trow + t * 16 + r + hi * 8)] =
                    (_Float16)acc[2][t][n][r];
            }
}

// ---------------------------------------------------------------------------
// Kernel 2: causal flash attention. Block = 4 waves = 64 query rows;
// 64-key K/V tiles, double-buffered async global->LDS staging.
// ---------------------------------------------------------------------------
__global__ __launch_bounds__(128) void attn_kernel(
    const _Float16* __restrict__ qh, const _Float16* __restrict__ kh,
    const _Float16* __restrict__ vT, float* __restrict__ out) {
    __shared__ __align__(16) _Float16 Kbuf[2][64 * KPITCH];  // [key][h]
    __shared__ __align__(16) _Float16 Vbuf[2][64 * KPITCH];  // [h][key]
    __shared__ __align__(16) _Float16 Ps[4][16 * PPITCH];    // per-wave P slab

    const int tid  = threadIdx.x;
    const int lane = tid & 31;
    const int w    = tid >> 5;

    const int blocksPerBatch = TSEQ / 64;
    const int b  = blockIdx.x / blocksPerBatch;
    const int qb = (blockIdx.x - b * blocksPerBatch) * 64;

    const _Float16* qB = qh + (size_t)b * TSEQ * HEAD;
    const _Float16* kB = kh + (size_t)b * TSEQ * HEAD;
    const _Float16* vB = vT + (size_t)b * HEAD * TSEQ;

    const int qr0    = qb + w * 16;
    const int arow   = lane & 15;
    const int khalf  = (lane >> 4) << 3;
    const int kstart = (lane >> 4) << 4;
    const int hi     = lane >> 4;
    const int n15    = lane & 15;

    // q A-operands for the two 32-wide head-dim chunks (loaded once)
    v16h Aq[2];
    #pragma unroll
    for (int hs = 0; hs < 2; ++hs) {
        const _Float16* qp = qB + (size_t)(qr0 + arow) * HEAD + hs * 32;
        Aq[hs] = ldA16(qp + kstart, qp + kstart + 8);
    }

    v8f O[4];
    #pragma unroll
    for (int n = 0; n < 4; ++n) O[n] = (v8f){0, 0, 0, 0, 0, 0, 0, 0};
    float mrow[8], lrow[8];
    #pragma unroll
    for (int r = 0; r < 8; ++r) { mrow[r] = -1e30f; lrow[r] = 0.f; }

    const float scale = 0.04419417382415922f;   // 512^-0.5 (ref scales by C)
    const int nkt = qb / 64 + 1;                // causal 64-key tiles

    // Stage tile t0 into buffer bufsel: 8 b128 async copies per thread.
    auto stage = [&](int t0, int bufsel) {
        _Float16* Kd = &Kbuf[bufsel][0];
        _Float16* Vd = &Vbuf[bufsel][0];
        const _Float16* ks = kB + (size_t)t0 * HEAD;
        const _Float16* vs = vB + t0;
        #pragma unroll
        for (int s = 0; s < 4; ++s) {
            int i = tid + s * 128;                      // 0..511
            int kk = i >> 3, c = (i & 7) * 8;           // key row, h chunk
            cp16_g2l(Kd + kk * KPITCH + c, ks + (size_t)kk * HEAD + c);
        }
        #pragma unroll
        for (int s = 0; s < 4; ++s) {
            int i = tid + s * 128;
            int h = i >> 3, c = (i & 7) * 8;            // h row, key chunk
            cp16_g2l(Vd + h * KPITCH + c, vs + (size_t)h * TSEQ + c);
        }
    };

    stage(0, 0);

    for (int kt = 0; kt < nkt; ++kt) {
        const int t0 = kt * 64;
        __syncthreads();                  // WAR: all readers of buf[(kt+1)&1] done
        if (kt + 1 < nkt) {
            stage(t0 + 64, (kt + 1) & 1); // overlap with this tile's compute
            wait_async_le8();             // tile kt's 8 copies have landed
        } else {
            wait_async_le0();
        }
        if (kt + 2 < nkt) {
            __builtin_prefetch(kB + (size_t)(t0 + 128) * HEAD, 0, 1);
            __builtin_prefetch(vB + t0 + 128, 0, 1);
        }
        __syncthreads();                  // everyone's tile-kt data visible

        const _Float16* Kb = &Kbuf[kt & 1][0];
        const _Float16* Vb = &Vbuf[kt & 1][0];

        // S = q @ k^T : four 16-key column tiles
        v8f S[4];
        #pragma unroll
        for (int j = 0; j < 4; ++j) {
            S[j] = (v8f){0, 0, 0, 0, 0, 0, 0, 0};
            #pragma unroll
            for (int hs = 0; hs < 2; ++hs) {
                v16h Bk = ldB16(Kb + (size_t)(j * 16 + n15) * KPITCH + hs * 32 + kstart);
                S[j] = __builtin_amdgcn_wmma_f32_16x16x32_f16(
                    false, Aq[hs], false, Bk, (short)0, S[j], false, false);
            }
        }

        // Scale + causal mask in C-layout (row = qr0+r+hi*8, col = t0+j*16+n15)
        #pragma unroll
        for (int j = 0; j < 4; ++j) {
            const int col = t0 + j * 16 + n15;
            #pragma unroll
            for (int r = 0; r < 8; ++r) {
                float s = S[j][r] * scale;
                if (col > qr0 + r + hi * 8) s = -1e30f;
                S[j][r] = s;
            }
        }

        // Online softmax per row; reductions stay inside 16-lane halves
        #pragma unroll
        for (int r = 0; r < 8; ++r) {
            float rm = fmaxf(fmaxf(S[0][r], S[1][r]), fmaxf(S[2][r], S[3][r]));
            rm = fmaxf(rm, __shfl_xor(rm, 1));
            rm = fmaxf(rm, __shfl_xor(rm, 2));
            rm = fmaxf(rm, __shfl_xor(rm, 4));
            rm = fmaxf(rm, __shfl_xor(rm, 8));
            const float mn = fmaxf(mrow[r], rm);
            float p[4], rs = 0.f;
            #pragma unroll
            for (int j = 0; j < 4; ++j) {
                p[j] = (S[j][r] < -1e29f) ? 0.f : __expf(S[j][r] - mn);
                rs += p[j];
            }
            rs += __shfl_xor(rs, 1);
            rs += __shfl_xor(rs, 2);
            rs += __shfl_xor(rs, 4);
            rs += __shfl_xor(rs, 8);
            const float corr = __expf(mrow[r] - mn);
            lrow[r] = lrow[r] * corr + rs;
            mrow[r] = mn;
            #pragma unroll
            for (int n = 0; n < 4; ++n) O[n][r] *= corr;
            const int prow = (r + hi * 8) * PPITCH;
            #pragma unroll
            for (int j = 0; j < 4; ++j)
                Ps[w][prow + j * 16 + n15] = (_Float16)p[j];
        }
        // wave-local LDS RAW: P writes -> A-layout reads (cross-lane)
        asm volatile("s_wait_dscnt 0" ::: "memory");

        v16h Ap[2];
        const _Float16* pp = &Ps[w][(size_t)arow * PPITCH];
        #pragma unroll
        for (int kc = 0; kc < 2; ++kc)
            Ap[kc] = ldA16(pp + kc * 32 + kstart, pp + kc * 32 + kstart + 8);

        // O += P @ V
        #pragma unroll
        for (int n = 0; n < 4; ++n) {
            #pragma unroll
            for (int kc = 0; kc < 2; ++kc) {
                v16h Bv = ldB16(Vb + (size_t)(n * 16 + n15) * KPITCH + kc * 32 + kstart);
                O[n] = __builtin_amdgcn_wmma_f32_16x16x32_f16(
                    false, Ap[kc], false, Bv, (short)0, O[n], false, false);
            }
        }
    }

    // Epilogue: divide by softmax denominator, store fp32 [B,T,H]
    #pragma unroll
    for (int r = 0; r < 8; ++r) {
        const float inv = 1.0f / lrow[r];
        const size_t row = (size_t)b * TSEQ + qr0 + r + hi * 8;
        #pragma unroll
        for (int n = 0; n < 4; ++n)
            out[row * HEAD + n * 16 + n15] = O[n][r] * inv;
    }
}

// ---------------------------------------------------------------------------
extern "C" void kernel_launch(void* const* d_in, const int* in_sizes, int n_in,
                              void* d_out, int out_size, void* d_ws, size_t ws_size,
                              hipStream_t stream) {
    (void)in_sizes; (void)n_in; (void)out_size; (void)ws_size;
    const float* x  = (const float*)d_in[0];
    const float* Wq = (const float*)d_in[1];
    const float* Wk = (const float*)d_in[2];
    const float* Wv = (const float*)d_in[3];
    float* out = (float*)d_out;

    const size_t nqkv = (size_t)BATCH * TSEQ * HEAD;  // halves per q/k/v
    _Float16* qh = (_Float16*)d_ws;
    _Float16* kh = qh + nqkv;
    _Float16* vT = kh + nqkv;
    _Float16* Wt = vT + nqkv;                          // 3*64*512 halves

    wt_kernel<<<(3 * HEAD * CEMB) / 256, 256, 0, stream>>>(Wq, Wk, Wv, Wt);
    proj_kernel<<<(BATCH * TSEQ) / 128, 128, 0, stream>>>(x, Wt, qh, kh, vT);
    attn_kernel<<<(BATCH * TSEQ) / 64, 128, 0, stream>>>(qh, kh, vT, out);
}
